// AttentionLayer_29540785062357
// MI455X (gfx1250) — matrix operand order repaired and checked by reading, not compiled, not run on hardware
//
#include <hip/hip_runtime.h>
#include <hip/hip_bf16.h>
#include <math.h>

// MI455X / gfx1250, wave32. Matrix op: V_WMMA_F32_16X16X32_F16 (probe-confirmed builtin).
typedef __attribute__((ext_vector_type(16))) _Float16 v16h;
typedef __attribute__((ext_vector_type(8)))  float    v8f;
typedef __attribute__((ext_vector_type(4)))  float    v4f;

#define S_LEN 4096
#define D_LEN 512
#define NBATCH 8

// ---------------------------------------------------------------------------
// Kernel 1: scores s[b][j] = <x[b,4095,:], x[b,j,:]> via WMMA.
// One wave handles a 16-key tile; A = 16 key rows (f16), B = q broadcast to
// all 16 columns. K=512 swept in chunks of 32 -> 16 chained v_wmma per tile,
// unrolled x4 so loads/cvts pipeline against the XDL matrix issues.
// grid = (8 batches, 32 tile-groups), block = 256 (8 waves * 16 keys = 128 keys).
// ---------------------------------------------------------------------------
__global__ __launch_bounds__(256) void scores_wmma(const float* __restrict__ X0,
                                                   float* __restrict__ s_ws) {
    const int batch = blockIdx.x;
    const int wave  = threadIdx.x >> 5;
    const int lane  = threadIdx.x & 31;
    const int tile  = blockIdx.y * 8 + wave;        // 0..255 -> key block j0 = tile*16
    const int row   = tile * 16 + (lane & 15);      // key row this lane feeds into A
    const bool hi   = (lane >= 16);

    const float* X = X0 + (size_t)batch * S_LEN * D_LEN;
    const float* q = X + (size_t)(S_LEN - 1) * D_LEN;   // last row = query

    // f16 A-matrix 16x32 layout: lane<16 covers K {0..7,16..23}, lane>=16 {8..15,24..31}
    const float* xr_base = X + (size_t)row * D_LEN + (hi ? 8 : 0);
    // f16 B-matrix 32x16 layout: lane<16 covers K {0..7,16..23}, lane>=16 {8..15,24..31}
    const float* qr_base = q + (hi ? 8 : 0);

    v8f acc = {};
#pragma unroll 4
    for (int k0 = 0; k0 < D_LEN; k0 += 32) {
        v16h amat, bmat;
        const float* xr = xr_base + k0;
        const float* qr = qr_base + k0;
#pragma unroll
        for (int e = 0; e < 8; ++e) {
            amat[e]     = (_Float16)xr[e];        // K = base + e
            amat[e + 8] = (_Float16)xr[e + 16];   // K = base + 16 + e
            bmat[e]     = (_Float16)qr[e];        // K = (hi?8:0) + e
            bmat[e + 8] = (_Float16)qr[e + 16];   // K = (hi?8:0) + 16 + e
        }
        acc = __builtin_amdgcn_wmma_f32_16x16x32_f16(
            /*neg_a=*/false, amat, /*neg_b=*/false, bmat,
            /*c_mod=*/(short)0, acc, /*reuse_a=*/false, /*reuse_b=*/false);
    }

    // All 16 D columns are identical (B was broadcast). C/D layout: lane group
    // 0-15 holds rows M=0..7 in regs 0..7, lanes 16-31 hold M=8..15.
    // Lane 0 emits s[j0..j0+7], lane 16 emits s[j0+8..j0+15] -> contiguous, so
    // spill each as two b128 stores instead of 8 scalar stores.
    if (lane == 0 || lane == 16) {
        float* dst = s_ws + batch * S_LEN + tile * 16 + (hi ? 8 : 0);
        v4f lo4 = { acc[0], acc[1], acc[2], acc[3] };
        v4f hi4 = { acc[4], acc[5], acc[6], acc[7] };
        *(v4f*)(dst)     = lo4;
        *(v4f*)(dst + 4) = hi4;
    }
}

// ---------------------------------------------------------------------------
// Kernel 2: per-batch softmax stats (rowmax, sum of exp). 8 blocks x 1024 thr.
// wave32 shuffle reduce + LDS cross-wave reduce.
// ---------------------------------------------------------------------------
__global__ __launch_bounds__(1024) void softmax_stats(const float* __restrict__ s_ws,
                                                      float* __restrict__ stats) {
    const int batch = blockIdx.x;
    const int tid   = threadIdx.x;
    const int wave  = tid >> 5;
    const int lane  = tid & 31;
    const float* sb = s_ws + batch * S_LEN;
    __shared__ float red[32];

    // --- max ---
    float m = -INFINITY;
    for (int i = tid; i < S_LEN; i += 1024) m = fmaxf(m, sb[i]);
#pragma unroll
    for (int off = 16; off; off >>= 1) m = fmaxf(m, __shfl_xor(m, off, 32));
    if (lane == 0) red[wave] = m;
    __syncthreads();
    if (tid < 32) {
        float v = red[tid];
#pragma unroll
        for (int off = 16; off; off >>= 1) v = fmaxf(v, __shfl_xor(v, off, 32));
        if (tid == 0) red[0] = v;
    }
    __syncthreads();
    const float mmax = red[0];
    __syncthreads();   // red[] is reused below

    // --- sum of exp ---
    float acc = 0.f;
    for (int i = tid; i < S_LEN; i += 1024) acc += expf(sb[i] - mmax);
#pragma unroll
    for (int off = 16; off; off >>= 1) acc += __shfl_xor(acc, off, 32);
    if (lane == 0) red[wave] = acc;
    __syncthreads();
    if (tid < 32) {
        float v = red[tid];
#pragma unroll
        for (int off = 16; off; off >>= 1) v += __shfl_xor(v, off, 32);
        if (tid == 0) {
            stats[2 * batch]     = mmax;
            stats[2 * batch + 1] = v;
        }
    }
}

// ---------------------------------------------------------------------------
// Kernel 3: partial context sums. grid (8 batches, 16 j-chunks) x 256 thr.
// Thread t owns output columns d = t and d = t+256 -> fully coalesced reads of X.
// Partials to workspace (no float atomics -> bitwise deterministic).
// ---------------------------------------------------------------------------
__global__ __launch_bounds__(256) void context_partial(const float* __restrict__ X0,
                                                       const float* __restrict__ s_ws,
                                                       const float* __restrict__ stats,
                                                       float* __restrict__ part) {
    const int batch = blockIdx.x;
    const int chunk = blockIdx.y;           // 16 chunks of 256 rows
    const int tid   = threadIdx.x;
    const float* Xb = X0 + (size_t)batch * S_LEN * D_LEN;
    const float  m     = stats[2 * batch];
    const float  denom = stats[2 * batch + 1];
    const int j0 = chunk * 256;

    __shared__ float w[256];
    w[tid] = expf(s_ws[batch * S_LEN + j0 + tid] - m) / denom;
    __syncthreads();

    float a0 = 0.f, a1 = 0.f;
    for (int jj = 0; jj < 256; ++jj) {
        const float* row = Xb + (size_t)(j0 + jj) * D_LEN;
        const float wj = w[jj];
        a0 += wj * row[tid];
        a1 += wj * row[tid + 256];
    }
    float* p = part + (size_t)(batch * 16 + chunk) * D_LEN;
    p[tid]       = a0;
    p[tid + 256] = a1;
}

// ---------------------------------------------------------------------------
// Kernel 4: deterministic reduction of the 16 partials -> d_out (writes every
// element, so the poisoned output buffer never needs a separate zero pass).
// ---------------------------------------------------------------------------
__global__ __launch_bounds__(256) void reduce_out(const float* __restrict__ part,
                                                  float* __restrict__ out) {
    const int idx = blockIdx.x * 256 + threadIdx.x;   // 0..4095
    const int batch = idx >> 9;                       // /512
    const int d     = idx & (D_LEN - 1);
    float acc = 0.f;
#pragma unroll
    for (int c = 0; c < 16; ++c)
        acc += part[(size_t)(batch * 16 + c) * D_LEN + d];
    out[idx] = acc;
}

// ---------------------------------------------------------------------------
extern "C" void kernel_launch(void* const* d_in, const int* in_sizes, int n_in,
                              void* d_out, int out_size, void* d_ws, size_t ws_size,
                              hipStream_t stream) {
    (void)in_sizes; (void)n_in; (void)out_size; (void)ws_size;
    const float* X = (const float*)d_in[0];   // [8, 4096, 512] f32
    float* out = (float*)d_out;               // [8, 512] f32

    // workspace layout (floats): scores [8*4096] | stats [16] | partials [8*16*512]
    float* s_ws  = (float*)d_ws;
    float* stats = s_ws + NBATCH * S_LEN;
    float* part  = stats + 2 * NBATCH;

    scores_wmma    <<<dim3(NBATCH, 32), 256,  0, stream>>>(X, s_ws);
    softmax_stats  <<<NBATCH,           1024, 0, stream>>>(s_ws, stats);
    context_partial<<<dim3(NBATCH, 16), 256,  0, stream>>>(X, s_ws, stats, part);
    reduce_out     <<<16,               256,  0, stream>>>(part, out);
}
